// UniPhyBlock_70643622085293
// MI455X (gfx1250) — matrix operand order repaired and checked by reading, not compiled, not run on hardware
//
#include <hip/hip_runtime.h>
#include <hip/hip_bf16.h>

typedef __attribute__((ext_vector_type(16))) _Float16 v16h;
typedef __attribute__((ext_vector_type(8)))  float    v8f;
typedef __attribute__((ext_vector_type(2)))  float    v2f;

#define T_    32
#define D_    32
#define HW_   4096
#define NPIX  131072
#define LN_EPSF 1e-5f

// ---------------- WMMA wrappers ----------------
__device__ __forceinline__ v8f wmma16(v16h a, v16h b, v8f c) {
  return __builtin_amdgcn_wmma_f32_16x16x32_f16(false, a, false, b, (short)0, c, false, false);
}
__device__ __forceinline__ v8f wmma32(v2f a, v2f b, v8f c) {
  return __builtin_amdgcn_wmma_f32_16x16x4_f32(false, a, false, b, (short)0, c, false, false);
}

// A fragment: 16x32 f16, row-major source, leading dim lda.
// K-run is contiguous (j=0..15 -> K=kg*16+j), matching the swizzled B tiles.
__device__ __forceinline__ v16h load_a16(const _Float16* A, int lda, int lane) {
  int m = lane & 15, kg = lane >> 4;
  v16h r;
#pragma unroll
  for (int j = 0; j < 16; ++j) {
    int K = kg * 16 + j;
    r[j] = A[m * lda + K];
  }
  return r;
}
// B fragment from pre-swizzled tile storage: [tile][lane][16 f16] contiguous.
__device__ __forceinline__ v16h load_b16s(const _Float16* tile, int lane) {
  return *(const v16h*)(tile + lane * 16);
}
// f32 WMMA fragments (16x16x4)
__device__ __forceinline__ v2f load_a32(const float* A, int lda, int lane) {
  int m = lane & 15, kb = (lane >> 4) * 2;
  v2f r; r.x = A[m * lda + kb]; r.y = A[m * lda + kb + 1]; return r;
}
__device__ __forceinline__ v2f load_b32(const float* B, int ldb, int lane, float s) {
  int n = lane & 15, kb = (lane >> 4) * 2;
  v2f r; r.x = s * B[kb * ldb + n]; r.y = s * B[(kb + 1) * ldb + n]; return r;
}

// Complex 32x32 @ 32x32 matmul via f32 WMMA. One full wave.
__device__ __forceinline__ void cmatmul32(const float* Are, const float* Aim, int lda,
                                          const float* Bre, const float* Bim, int ldb,
                                          float* Cre, float* Cim, int ldc, int lane) {
#pragma unroll
  for (int mt = 0; mt < 2; ++mt) {
#pragma unroll
    for (int nt = 0; nt < 2; ++nt) {
      v8f ar8 = {0.f,0.f,0.f,0.f,0.f,0.f,0.f,0.f};
      v8f ai8 = {0.f,0.f,0.f,0.f,0.f,0.f,0.f,0.f};
      const float* Ar = Are + mt * 16 * lda;
      const float* Ai = Aim + mt * 16 * lda;
      const float* Br = Bre + nt * 16;
      const float* Bi = Bim + nt * 16;
#pragma unroll
      for (int ks = 0; ks < 8; ++ks) {
        v2f far = load_a32(Ar + ks * 4, lda, lane);
        v2f fai = load_a32(Ai + ks * 4, lda, lane);
        v2f fbr = load_b32(Br + ks * 4 * ldb, ldb, lane,  1.f);
        v2f fbi = load_b32(Bi + ks * 4 * ldb, ldb, lane,  1.f);
        v2f fbin = load_b32(Bi + ks * 4 * ldb, ldb, lane, -1.f);
        ar8 = wmma32(far, fbr,  ar8);   // + Re*Re
        ar8 = wmma32(fai, fbin, ar8);   // - Im*Im
        ai8 = wmma32(far, fbi,  ai8);   // + Re*Im
        ai8 = wmma32(fai, fbr,  ai8);   // + Im*Re
      }
      int n = lane & 15, m0 = (lane >> 4) * 8;
#pragma unroll
      for (int r = 0; r < 8; ++r) {
        Cre[(mt * 16 + m0 + r) * ldc + nt * 16 + n] = ar8[r];
        Cim[(mt * 16 + m0 + r) * ldc + nt * 16 + n] = ai8[r];
      }
    }
  }
}

__device__ __forceinline__ float gelu_tanh(float x) {
  float x3 = x * x * x;
  return 0.5f * x * (1.f + tanhf(0.79788456f * (x + 0.044715f * x3)));
}

// ------- K0: weight repack (into WMMA B-fragment lane order) + SSM operators -------
// wkts: conv weights, tiles [s(18)][ocTile(4)] of 32x16, element [lane][j]:
//   oc = ocTile*16 + (lane&15), K = s*32 + (lane>>4)*16 + j -> tap=K>>6, ic=K&63
// w1s : [e(4)][nt(8)][ks(2)] tiles of w1[e] (64x128)
// w2s : [e(4)][nt2(4)][ks(4)] tiles of w2[e] (128x64)
__global__ void prep_kernel(const float* conv_w, const float* w1, const float* w2,
                            const float* lam_re, const float* lam_im, const float* dt,
                            _Float16* wkts, _Float16* w1s, _Float16* w2s, float* ops) {
  int i = blockIdx.x * blockDim.x + threadIdx.x;
  if (i < 36864) {
    int j = i & 15, lane = (i >> 4) & 31, tile = (i >> 9) & 3, s = i >> 11;
    int oc = tile * 16 + (lane & 15);
    int kk = s * 32 + ((lane >> 4) & 1) * 16 + j;
    int tap = kk >> 6, ic = kk & 63;
    wkts[i] = (_Float16)conv_w[(oc * 64 + ic) * 9 + tap];
  } else if (i < 36864 + 32768) {
    int q = i - 36864;
    int j = q & 15, lane = (q >> 4) & 31, ks = (q >> 9) & 1, nt = (q >> 10) & 7, e = q >> 13;
    int k = ks * 32 + ((lane >> 4) & 1) * 16 + j;
    int n = nt * 16 + (lane & 15);
    w1s[q] = (_Float16)w1[e * 8192 + k * 128 + n];
  } else if (i < 36864 + 65536) {
    int q = i - 36864 - 32768;
    int j = q & 15, lane = (q >> 4) & 31, ks = (q >> 9) & 3, nt2 = (q >> 11) & 3, e = q >> 13;
    int k = ks * 32 + ((lane >> 4) & 1) * 16 + j;
    int n = nt2 * 16 + (lane & 15);
    w2s[q] = (_Float16)w2[e * 8192 + k * 64 + n];
  } else if (i < 36864 + 65536 + 1024) {
    int j = i - 36864 - 65536;
    int t = j >> 5, d = j & 31;
    float x  = lam_re[d];
    float sp = fmaxf(x, 0.f) + log1pf(expf(-fabsf(x)));
    float lr = -sp, li = lam_im[d];
    float dtc = dt[t];
    float er = expf(lr * dtc);
    float dr = er * cosf(li * dtc);
    float di = er * sinf(li * dtc);
    float den = lr * lr + li * li;
    float nr = dr - 1.f, ni = di;
    ops[j]        = dr;                        // decay re
    ops[1024 + j] = di;                        // decay im
    ops[2048 + j] = (nr * lr + ni * li) / den; // forcing re
    ops[3072 + j] = (ni * lr - nr * li) / den; // forcing im
  }
}

// ---------------- K1: LayerNorm-1, channel-last f16 output ----------------
__global__ void ln1_kernel(const float* x_re, const float* x_im,
                           const float* g, const float* b, _Float16* xn) {
  int pix = blockIdx.x * blockDim.x + threadIdx.x;
  if (pix >= NPIX) return;
  int t = pix >> 12, hw = pix & 4095;
  float s = 0.f, s2 = 0.f;
#pragma unroll 4
  for (int d = 0; d < 32; ++d) {
    float a = x_re[(t * 32 + d) * HW_ + hw];
    float c = x_im[(t * 32 + d) * HW_ + hw];
    s += a + c; s2 += a * a + c * c;
  }
  float m = s * (1.f / 64.f);
  float var = s2 * (1.f / 64.f) - m * m;
  float inv = rsqrtf(var + LN_EPSF);
  long base = (long)pix * 64;
#pragma unroll 4
  for (int d = 0; d < 32; ++d) {
    float a = x_re[(t * 32 + d) * HW_ + hw];
    float c = x_im[(t * 32 + d) * HW_ + hw];
    xn[base + d]      = (_Float16)((a - m) * inv * g[d]      + b[d]);
    xn[base + 32 + d] = (_Float16)((c - m) * inv * g[32 + d] + b[32 + d]);
  }
}

// ---------------- K2: 3x3 conv as implicit GEMM (f16 WMMA) ----------------
// Halo staging: in-bounds 16B chunks via global_load_async_to_lds_b128 (ASYNCcnt),
// out-of-bounds chunks zero-filled with plain ds stores (disjoint addresses).
__global__ void conv_kernel(const _Float16* xn, const _Float16* wkts,
                            const float* conv_b, float* xs) {
  __shared__ _Float16 a_lds[3 * 18 * 64];   // 3 rows x 18 px halo x 64 ch
  int tid = threadIdx.x;
  int lane = tid & 31, wave = tid >> 5;
  int pix_base = blockIdx.x * 16;
  int t = pix_base >> 12;
  int rem = pix_base & 4095;
  int h = rem >> 6, w0 = rem & 63;
  // 54 halo pixels x 8 chunks of 8 f16 (16B) = 432 chunks
  for (int idx = tid; idx < 432; idx += 128) {
    int pl = idx >> 3, chunk = idx & 7;
    int ry = pl / 18;
    int px = pl - ry * 18;
    int gh = h + ry - 1, gw = w0 + px - 1;
    int loff = ((ry * 18 + px) * 64 + chunk * 8);            // halfword offset
    if (gh >= 0 && gh < 64 && gw >= 0 && gw < 64) {
      const _Float16* gsrc = xn + ((long)(t * HW_ + gh * 64 + gw)) * 64 + chunk * 8;
      unsigned ldsa = (unsigned)(size_t)&a_lds[loff];
      asm volatile("global_load_async_to_lds_b128 %0, %1, off"
                   :: "v"(ldsa), "v"(gsrc) : "memory");
    } else {
      *(int4*)&a_lds[loff] = make_int4(0, 0, 0, 0);
    }
  }
  asm volatile("s_wait_asynccnt 0x0" ::: "memory");
  __syncthreads();

  v8f acc = {0.f,0.f,0.f,0.f,0.f,0.f,0.f,0.f};
#pragma unroll
  for (int s = 0; s < 18; ++s) {           // K = 9 taps x 64 ch = 18 chunks of 32
    int tap = s >> 1, half = s & 1;
    int kh = tap / 3, kw = tap - kh * 3;
    v16h a = load_a16(&a_lds[(kh * 18 + kw) * 64 + half * 32], 64, lane);
    v16h b = load_b16s(&wkts[((s * 4 + wave) * 32) * 16], lane);
    acc = wmma16(a, b, acc);
  }
  int n = lane & 15, m0 = (lane >> 4) * 8;
  int oc = wave * 16 + n;
  float bias = conv_b[oc];
#pragma unroll
  for (int r = 0; r < 8; ++r) {
    int pix = pix_base + m0 + r;
    xs[(long)pix * 64 + oc] = acc[r] + bias;
  }
}

// ---------------- K3: x_mean reduction over H*W ----------------
__global__ void xmean_reduce(const float* xs, float* xmean) {
  __shared__ float red[256];
  int bid = blockIdx.x;                    // t*64 + c
  int t = bid >> 6, c = bid & 63;
  float s = 0.f;
  for (int p = threadIdx.x; p < 4096; p += 256)
    s += xs[((long)((t << 12) + p)) * 64 + c];
  red[threadIdx.x] = s;
  __syncthreads();
  for (int off = 128; off > 0; off >>= 1) {
    if (threadIdx.x < off) red[threadIdx.x] += red[threadIdx.x + off];
    __syncthreads();
  }
  if (threadIdx.x == 0) xmean[bid] = red[0] * (1.f / 4096.f);
}

// ---------------- K4: flux GEMM (f32 WMMA) + sequential scan ----------------
__global__ void flux_kernel(const float* xmean, const float* flux_alpha,
                            const float* flux_wre, const float* flux_wim,
                            float* out_tail) {
  __shared__ float Mre[32 * 33], Mim[32 * 33], Xr[32 * 33], Xi[32 * 33];
  __shared__ float Wr[1024], Wi[1024];
  int lane = threadIdx.x;
  for (int i = lane; i < 1024; i += 32) { Wr[i] = flux_wre[i]; Wi[i] = flux_wim[i]; }
  for (int t = 0; t < 32; ++t) {
    Mre[t * 33 + lane] = xmean[t * 64 + lane];
    Mim[t * 33 + lane] = xmean[t * 64 + 32 + lane];
  }
  __syncthreads();
  cmatmul32(Mre, Mim, 33, Wr, Wi, 32, Xr, Xi, 33, lane);
  __syncthreads();
  float A = 1.f / (1.f + expf(-flux_alpha[lane]));
  float sr = 0.f, si = 0.f;
  for (int t = 0; t < 32; ++t) {
    sr = A * sr + Xr[t * 33 + lane];
    si = A * si + Xi[t * 33 + lane];
  }
  out_tail[lane]      = sr;
  out_tail[32 + lane] = si;
}

// ------- K5: per-pixel encode GEMM -> SSM scan -> noise -> decode -> LN2 -------
__global__ void seq_kernel(const float* xs,
                           const float* enc_re, const float* enc_im,
                           const float* dec_re, const float* dec_im,
                           const float* ops, const float* dt,
                           const float* eps_re, const float* eps_im,
                           const float* ln2g, const float* ln2b,
                           float* xon, _Float16* xonh) {
  __shared__ float Xre[32 * 33], Xim[32 * 33];   // input rows, later Y rows
  __shared__ float Ure[32 * 33], Uim[32 * 33];   // u, later x_out
  __shared__ float Er[1024], Ei[1024];
  int lane = threadIdx.x;
  int hw = blockIdx.x;

  for (int i = lane; i < 1024; i += 32) { Er[i] = enc_re[i]; Ei[i] = enc_im[i]; }
  for (int t = 0; t < 32; ++t) {
    long base = ((long)((t << 12) + hw)) * 64;
    Xre[t * 33 + lane] = xs[base + lane];
    Xim[t * 33 + lane] = xs[base + 32 + lane];
  }
  __syncthreads();
  cmatmul32(Xre, Xim, 33, Er, Ei, 32, Ure, Uim, 33, lane);   // u = xc @ enc
  __syncthreads();

  {
    int d = lane;
    float sr = 0.f, si = 0.f;
    for (int t = 0; t < 32; ++t) {
      int j = t * 32 + d;
      float fr = ops[2048 + j], fi = ops[3072 + j];
      float dr = ops[j],        di = ops[1024 + j];
      float ur = Ure[t * 33 + d], ui = Uim[t * 33 + d];
      float usr = ur * fr - ui * fi;
      float usi = ur * fi + ui * fr;
      float nsr = dr * sr - di * si + usr;
      float nsi = dr * si + di * sr + usi;
      sr = nsr; si = nsi;
      float ct = 0.01f * sqrtf(dt[t]);
      long eb = ((long)((t << 12) + hw)) * 32 + d;
      float er = eps_re[eb], ei = eps_im[eb];
      Xre[t * 33 + d] = sr + ct * (sr * er - si * ei);
      Xim[t * 33 + d] = si + ct * (sr * ei + si * er);
    }
  }
  __syncthreads();
  for (int i = lane; i < 1024; i += 32) { Er[i] = dec_re[i]; Ei[i] = dec_im[i]; }
  __syncthreads();
  cmatmul32(Xre, Xim, 33, Er, Ei, 32, Ure, Uim, 33, lane);   // x_out = Y' @ dec
  __syncthreads();

  {  // LayerNorm-2 over 64 = [re(32)|im(32)] per (t,pixel); lane = t
    int t = lane;
    float s = 0.f, s2 = 0.f;
    for (int c = 0; c < 32; ++c) {
      float a = Ure[t * 33 + c], b = Uim[t * 33 + c];
      s += a + b; s2 += a * a + b * b;
    }
    float m = s * (1.f / 64.f);
    float var = s2 * (1.f / 64.f) - m * m;
    float inv = rsqrtf(var + LN_EPSF);
    long base = ((long)((t << 12) + hw)) * 64;
    for (int c = 0; c < 32; ++c) {
      float a = (Ure[t * 33 + c] - m) * inv * ln2g[c] + ln2b[c];
      float b = (Uim[t * 33 + c] - m) * inv * ln2g[32 + c] + ln2b[32 + c];
      xon[base + c]       = a;  xonh[base + c]       = (_Float16)a;
      xon[base + 32 + c]  = b;  xonh[base + 32 + c]  = (_Float16)b;
    }
  }
}

// ------- K6: gated 4-expert MLP (f16 WMMA) + residual, writes final output -------
__global__ void moe_kernel(const float* xon, const _Float16* xonh,
                           const float* wg, const _Float16* w1s, const float* b1,
                           const _Float16* w2s, const float* b2,
                           const float* x_re, const float* x_im, float* out) {
  __shared__ _Float16 Af[16 * 64];
  __shared__ _Float16 Hdn[16 * 128];
  __shared__ float gate[64];
  int tid = threadIdx.x;
  int lane = tid & 31, wave = tid >> 5;
  int pix_base = blockIdx.x * 16;

  for (int i = tid; i < 1024; i += 128) Af[i] = xonh[(long)pix_base * 64 + i];
  if (tid < 64) {                       // gate logits
    int m = tid >> 2, e = tid & 3;
    float acc = 0.f;
    for (int c = 0; c < 64; ++c)
      acc += xon[((long)(pix_base + m)) * 64 + c] * wg[c * 4 + e];
    gate[m * 4 + e] = acc;
  }
  __syncthreads();
  if (tid < 16) {                       // softmax over 4 experts
    float l0 = gate[tid*4], l1 = gate[tid*4+1], l2 = gate[tid*4+2], l3 = gate[tid*4+3];
    float mx = fmaxf(fmaxf(l0, l1), fmaxf(l2, l3));
    float e0 = expf(l0-mx), e1 = expf(l1-mx), e2 = expf(l2-mx), e3 = expf(l3-mx);
    float inv = 1.f / (e0 + e1 + e2 + e3);
    gate[tid*4] = e0*inv; gate[tid*4+1] = e1*inv; gate[tid*4+2] = e2*inv; gate[tid*4+3] = e3*inv;
  }
  __syncthreads();

  float delta[8];
#pragma unroll
  for (int r = 0; r < 8; ++r) delta[r] = 0.f;
  int n = lane & 15, m0 = (lane >> 4) * 8;

  for (int e = 0; e < 4; ++e) {
    __builtin_prefetch(&w2s[(((e * 4 + wave) * 4) * 32 + lane) * 16], 0, 1);
#pragma unroll
    for (int sub = 0; sub < 2; ++sub) {        // w1: 8 N-tiles of FH=128, 2/wave
      int nt = wave * 2 + sub;
      v8f acc = {0.f,0.f,0.f,0.f,0.f,0.f,0.f,0.f};
#pragma unroll
      for (int ks = 0; ks < 2; ++ks) {
        v16h a = load_a16(Af + ks * 32, 64, lane);
        v16h b = load_b16s(&w1s[(((e * 8 + nt) * 2 + ks) * 32) * 16], lane);
        acc = wmma16(a, b, acc);
      }
#pragma unroll
      for (int r = 0; r < 8; ++r) {
        float h = acc[r] + b1[e * 128 + nt * 16 + n];
        Hdn[(m0 + r) * 128 + nt * 16 + n] = (_Float16)gelu_tanh(h);
      }
    }
    __syncthreads();
    {                                          // w2: wave owns 16 output channels
      v8f acc2 = {0.f,0.f,0.f,0.f,0.f,0.f,0.f,0.f};
#pragma unroll
      for (int ks = 0; ks < 4; ++ks) {
        v16h a = load_a16(Hdn + ks * 32, 128, lane);
        v16h b = load_b16s(&w2s[(((e * 4 + wave) * 4 + ks) * 32) * 16], lane);
        acc2 = wmma16(a, b, acc2);
      }
      int c = wave * 16 + n;
      float bb = b2[e * 64 + c];
#pragma unroll
      for (int r = 0; r < 8; ++r)
        delta[r] += gate[(m0 + r) * 4 + e] * (acc2[r] + bb);
    }
    __syncthreads();
  }

  // residual: out = x + xoc + delta
  int c = wave * 16 + n;
#pragma unroll
  for (int r = 0; r < 8; ++r) {
    int pix = pix_base + m0 + r;
    int t = pix >> 12, hw = pix & 4095;
    float xo = xon[(long)pix * 64 + c];
    if (c < 32) {
      long idx = (long)(t * 32 + c) * HW_ + hw;
      out[idx] = x_re[idx] + xo + delta[r];
    } else {
      long idx = (long)(t * 32 + (c - 32)) * HW_ + hw;
      out[4194304 + idx] = x_im[idx] + xo + delta[r];
    }
  }
}

// ---------------- host launcher ----------------
extern "C" void kernel_launch(void* const* d_in, const int* in_sizes, int n_in,
                              void* d_out, int out_size, void* d_ws, size_t ws_size,
                              hipStream_t stream) {
  const float* x_re    = (const float*)d_in[0];
  const float* x_im    = (const float*)d_in[1];
  const float* dt      = (const float*)d_in[2];
  const float* eps_re  = (const float*)d_in[3];
  const float* eps_im  = (const float*)d_in[4];
  const float* ln1_g   = (const float*)d_in[5];
  const float* ln1_b   = (const float*)d_in[6];
  const float* ln2_g   = (const float*)d_in[7];
  const float* ln2_b   = (const float*)d_in[8];
  const float* conv_w  = (const float*)d_in[9];
  const float* conv_b  = (const float*)d_in[10];
  const float* enc_re  = (const float*)d_in[11];
  const float* enc_im  = (const float*)d_in[12];
  const float* dec_re  = (const float*)d_in[13];
  const float* dec_im  = (const float*)d_in[14];
  const float* lam_re  = (const float*)d_in[15];
  const float* lam_im  = (const float*)d_in[16];
  const float* flux_a  = (const float*)d_in[17];
  const float* flux_wre= (const float*)d_in[18];
  const float* flux_wim= (const float*)d_in[19];
  const float* wg      = (const float*)d_in[20];
  const float* w1      = (const float*)d_in[21];
  const float* b1      = (const float*)d_in[22];
  const float* w2      = (const float*)d_in[23];
  const float* b2      = (const float*)d_in[24];
  float* out = (float*)d_out;

  char* ws = (char*)d_ws;
  _Float16* xn    = (_Float16*)(ws + 0);               // 16 MiB
  float*    xs    = (float*)   (ws + 16777216);        // 32 MiB
  float*    xon   = (float*)   (ws + 50331648);        // 32 MiB
  _Float16* xonh  = (_Float16*)(ws + 83886080);        // 16 MiB
  _Float16* wkts  = (_Float16*)(ws + 100663296);       // 72 KiB (swizzled)
  _Float16* w1s   = (_Float16*)(ws + 100737024);       // 64 KiB (swizzled)
  _Float16* w2s   = (_Float16*)(ws + 100802560);       // 64 KiB (swizzled)
  float*    xmean = (float*)   (ws + 100868096);       // 8 KiB
  float*    ops   = (float*)   (ws + 100876288);       // 16 KiB

  prep_kernel<<<404, 256, 0, stream>>>(conv_w, w1, w2, lam_re, lam_im, dt,
                                       wkts, w1s, w2s, ops);
  ln1_kernel<<<512, 256, 0, stream>>>(x_re, x_im, ln1_g, ln1_b, xn);
  conv_kernel<<<8192, 128, 0, stream>>>(xn, wkts, conv_b, xs);
  xmean_reduce<<<2048, 256, 0, stream>>>(xs, xmean);
  flux_kernel<<<1, 32, 0, stream>>>(xmean, flux_a, flux_wre, flux_wim, out + 8388608);
  seq_kernel<<<4096, 32, 0, stream>>>(xs, enc_re, enc_im, dec_re, dec_im,
                                      ops, dt, eps_re, eps_im, ln2_g, ln2_b,
                                      xon, xonh);
  moe_kernel<<<8192, 128, 0, stream>>>(xon, xonh, wg, w1s, b1, w2s, b2,
                                       x_re, x_im, out);
}
